// MotifModel_lstm_50173807952816
// MI455X (gfx1250) — matrix operand order repaired and checked by reading, not compiled, not run on hardware
//
#include <hip/hip_runtime.h>
#include <hip/hip_bf16.h>

// ---------------- model dims ----------------
#define BB   64
#define LL   2048
#define HH   64
#define NL   10
#define NRES 3
#define EPSV 1e-5f

typedef __attribute__((ext_vector_type(16))) _Float16 v16h;
typedef __attribute__((ext_vector_type(8)))  _Float16 v8h;
typedef __attribute__((ext_vector_type(8)))  float    v8f;

// ---------- CDNA5 async global->LDS helpers (ASYNCcnt path) ----------
// Each active lane copies 16B from its global address to its LDS offset.
__device__ __forceinline__ void async_g2l_b128(const void* gsrc, void* ldst) {
    asm volatile("global_load_async_to_lds_b128 %0, %1, off"
                 :: "v"((unsigned)(uintptr_t)ldst),
                    "v"((unsigned long long)(uintptr_t)gsrc)
                 : "memory");
}
__device__ __forceinline__ void wait_async0() {
    asm volatile("s_wait_asynccnt 0x0" ::: "memory");
}

// ---------- WMMA fragment helpers (16x16x32 f16, wave32) ----------
// A-matrix 16x32: lane M = lane&15; per-lane data = 16 contiguous halves
// at K-offsets [group*16, +16).
__device__ __forceinline__ v16h frag_a_contig(const _Float16* p, int group) {
    v8h x = *(const v8h*)(p + group * 16);
    v8h y = *(const v8h*)(p + group * 16 + 8);
    return __builtin_shufflevector(x, y, 0, 1, 2, 3, 4, 5, 6, 7,
                                   8, 9, 10, 11, 12, 13, 14, 15);
}

// B-matrix 32x16: lane N = lane&15; lanes 0-15 hold K=0..15, 16-31 hold 16..31.
// Per-lane data = 16 contiguous halves (32B).
__device__ __forceinline__ v16h frag_b16(const _Float16* p) {
    v8h x = *(const v8h*)p;
    v8h y = *(const v8h*)(p + 8);
    return __builtin_shufflevector(x, y, 0, 1, 2, 3, 4, 5, 6, 7,
                                   8, 9, 10, 11, 12, 13, 14, 15);
}

// ================= front: conv1(4->32,k1)+relu, conv2(32->64,k1)+relu =================
__global__ __launch_bounds__(256) void front_kernel(
    const float* __restrict__ in,     // [B,L,4]
    const float* __restrict__ w1, const float* __restrict__ b1,   // [32,4],[32]
    const float* __restrict__ w2, const float* __restrict__ b2,   // [64,32],[64]
    float* __restrict__ X32)          // [B,64,L]
{
    int idx = blockIdx.x * blockDim.x + threadIdx.x;
    if (idx >= BB * LL) return;
    int b = idx >> 11, l = idx & (LL - 1);
    float x4[4];
#pragma unroll
    for (int i = 0; i < 4; ++i) x4[i] = in[((size_t)b * LL + l) * 4 + i];
    float t32[32];
#pragma unroll 4
    for (int o = 0; o < 32; ++o) {
        float s = b1[o];
#pragma unroll
        for (int i = 0; i < 4; ++i) s += w1[o * 4 + i] * x4[i];
        t32[o] = fmaxf(s, 0.f);
    }
#pragma unroll 2
    for (int o = 0; o < 64; ++o) {
        float s = b2[o];
#pragma unroll
        for (int j = 0; j < 32; ++j) s += w2[o * 32 + j] * t32[j];
        X32[((size_t)b * 64 + o) * LL + l] = fmaxf(s, 0.f);
    }
}

// ========== elementwise BN + ReLU, [B,64,L] f32 -> channels-last [B,L,64] f16 ==========
__global__ __launch_bounds__(256) void bnrelu_cl_kernel(
    const float* __restrict__ x, const float* __restrict__ g, const float* __restrict__ bb,
    const float* __restrict__ m, const float* __restrict__ v, _Float16* __restrict__ o, int n)
{
    int idx = blockIdx.x * blockDim.x + threadIdx.x;   // read-order index [b][c][l]
    if (idx >= n) return;
    int l = idx & (LL - 1);
    int c = (idx >> 11) & 63;
    int b = idx >> 17;
    float s = g[c] * rsqrtf(v[c] + EPSV);
    float val = (x[idx] - m[c]) * s + bb[c];
    o[((size_t)b << 17) + ((size_t)l << 6) + c] = (_Float16)fmaxf(val, 0.f);
}

// ============ repack res-conv weights: f32 [oc][i][k] -> f16 [oc][k][i] ============
__global__ __launch_bounds__(256) void wrepack_kernel(const float* __restrict__ in,
                                                      _Float16* __restrict__ out, int nmat)
{
    int idx = blockIdx.x * blockDim.x + threadIdx.x;   // out-order index
    if (idx >= nmat * 45056) return;
    int m = idx / 45056, rem = idx - m * 45056;
    int oc = rem / 704, r2 = rem - oc * 704;
    int k = r2 >> 6, i = r2 & 63;
    out[idx] = (_Float16)in[(size_t)m * 45056 + ((size_t)oc * 64 + i) * 11 + k];
}

// ================= residual conv k=11 (64->64) via WMMA =================
// Block = 128 threads = 4 waves; block covers (b, 64 positions); wave w owns
// output-channel tile m0=16w across 4 N-subtiles. Input stripe (channels-last)
// staged via async global->LDS DMA; A/B-frags are pure b128 loads.
__global__ __launch_bounds__(128) void conv11_wmma_kernel(
    const _Float16* __restrict__ hcl,  // [B,L,64] f16 channels-last
    const _Float16* __restrict__ w16,  // [64,11,64] f16 repacked
    const float* __restrict__ bias,    // [64]
    float* __restrict__ out,           // [B,64,L] f32 : write or accumulate
    int accum)
{
    __shared__ __align__(16) _Float16 sIn[80 * 64];    // [pos l0-8+p][ch]

    int tid = threadIdx.x;
    int wave = tid >> 5, lane = tid & 31;
    int group = lane >> 4, nlo = lane & 15;
    int l0 = blockIdx.x * 64;
    int b  = blockIdx.y;
    int m0 = wave * 16;

    // stage 80 positions x 64 ch, 16B chunks; in-range rows via async DMA
    for (int i8 = tid; i8 < 80 * 8; i8 += 128) {
        int p = i8 >> 3, c8 = (i8 & 7) << 3;
        int l = l0 - 8 + p;
        if (l >= 0 && l < LL) {
            async_g2l_b128(&hcl[(((size_t)b * LL + l) << 6) + c8], &sIn[p * 64 + c8]);
        } else {
#pragma unroll
            for (int e = 0; e < 8; ++e) sIn[p * 64 + c8 + e] = (_Float16)0.f;
        }
    }
    wait_async0();
    __syncthreads();

    v8f acc[4];
#pragma unroll
    for (int nt = 0; nt < 4; ++nt)
#pragma unroll
        for (int r = 0; r < 8; ++r) acc[nt][r] = bias[m0 + 8 * group + r];

    for (int k = 0; k < 11; ++k) {
#pragma unroll
        for (int kk = 0; kk < 2; ++kk) {
            // A: weights [oc][k][i]: contiguous over i -> two b128 loads
            const _Float16* wp = w16 + ((size_t)(m0 + nlo) * 11 + k) * 64 + kk * 32;
            v16h a = frag_a_contig(wp, group);
            // B from LDS: pos p = nt*16 + nlo + k + 3, ch = kk*32 + group*16 + e
#pragma unroll
            for (int nt = 0; nt < 4; ++nt) {
                v16h bf = frag_b16(&sIn[(nt * 16 + nlo + k + 3) * 64 + kk * 32 + group * 16]);
                acc[nt] = __builtin_amdgcn_wmma_f32_16x16x32_f16(false, a, false, bf,
                                                                 (short)0, acc[nt], false, false);
            }
        }
    }
#pragma unroll
    for (int nt = 0; nt < 4; ++nt) {
#pragma unroll
        for (int r = 0; r < 8; ++r) {
            int oc = m0 + 8 * group + r;
            size_t idx = ((size_t)b * 64 + oc) * LL + l0 + nt * 16 + nlo;
            if (accum) out[idx] += acc[nt][r];
            else       out[idx]  = acc[nt][r];
        }
    }
}

// ================= [B,64,L] f32 -> [L,B,64] f16 (seq-first) =================
__global__ __launch_bounds__(256) void to_seq_kernel(const float* __restrict__ X32,
                                                     _Float16* __restrict__ S)
{
    int idx = blockIdx.x * blockDim.x + threadIdx.x;
    if (idx >= BB * 64 * LL) return;
    int b = idx / (64 * LL);
    int rem = idx - b * (64 * LL);
    int c = rem >> 11, l = rem & (LL - 1);
    S[((size_t)l * BB + b) * HH + c] = (_Float16)X32[idx];
}

// ================= f32 -> f16 convert =================
__global__ __launch_bounds__(256) void cvt_kernel(const float* __restrict__ in,
                                                  _Float16* __restrict__ out, int n)
{
    int idx = blockIdx.x * blockDim.x + threadIdx.x;
    if (idx < n) out[idx] = (_Float16)in[idx];
}

// ================= one full LSTM layer, resident in one workgroup =================
// gates[b][g] = bias[g] + sum_h X_t[b][h]*Wih[g][h] + sum_h Hprev[b][h]*Whh[g][h]
// M = batch (64 -> 4 tiles), N = gates (256 -> 16 tiles), K = 64 (2 chunks)
__global__ __launch_bounds__(256) void lstm_layer_kernel(
    const _Float16* __restrict__ Sin,   // [T,B,H] f16
    _Float16* __restrict__ Sout,        // [T,B,H] f16
    const _Float16* __restrict__ Wih,   // [256,64] f16
    const _Float16* __restrict__ Whh,   // [256,64] f16
    const float* __restrict__ bih, const float* __restrict__ bhh)  // [256]
{
    __shared__ __align__(16) _Float16 sWih[256 * 64];
    __shared__ __align__(16) _Float16 sWhh[256 * 64];
    __shared__ __align__(16) _Float16 sX[64 * 64];
    __shared__ __align__(16) _Float16 sH[64 * 64];
    __shared__ __align__(16) float    sC[64 * 64];
    __shared__ __align__(16) _Float16 sG[64 * 256];
    __shared__ __align__(16) float    sBias[256];

    int tid = threadIdx.x;

    // weights -> LDS via CDNA5 async global->LDS DMA (16B per lane per issue)
#pragma unroll
    for (int i = tid * 8; i < 256 * 64; i += 256 * 8) {
        async_g2l_b128(Wih + i, &sWih[i]);
        async_g2l_b128(Whh + i, &sWhh[i]);
    }
    if (tid < 256) sBias[tid] = bih[tid] + bhh[tid];
    for (int i = tid; i < 64 * 64; i += 256) { sH[i] = (_Float16)0.f; sC[i] = 0.f; }
    wait_async0();
    __syncthreads();

    int wave = tid >> 5, lane = tid & 31;
    int group = lane >> 4, nlo = lane & 15;

    for (int t = 0; t < LL; ++t) {
        // stage X_t (8KB) into LDS via async DMA, then fence with ASYNCcnt
        const _Float16* xin = Sin + (size_t)t * BB * HH;
        async_g2l_b128(xin + tid * 8,        &sX[tid * 8]);
        async_g2l_b128(xin + 2048 + tid * 8, &sX[2048 + tid * 8]);
        if (t + 1 < LL)
            __builtin_prefetch(Sin + (size_t)(t + 1) * BB * HH + tid * 16, 0, 0);
        wait_async0();
        __syncthreads();

        // 64 gate tiles: wave w owns N-tiles {2w, 2w+1} x 4 M-tiles
#pragma unroll
        for (int j = 0; j < 2; ++j) {
            int g0 = (wave * 2 + j) * 16;
            const _Float16* wihrow = sWih + (size_t)(g0 + nlo) * 64 + group * 16;
            const _Float16* whhrow = sWhh + (size_t)(g0 + nlo) * 64 + group * 16;
            v16h bih0 = frag_b16(wihrow);
            v16h bih1 = frag_b16(wihrow + 32);
            v16h bhh0 = frag_b16(whhrow);
            v16h bhh1 = frag_b16(whhrow + 32);
            float bval = sBias[g0 + nlo];
#pragma unroll
            for (int mt = 0; mt < 4; ++mt) {
                v8f acc;
#pragma unroll
                for (int r = 0; r < 8; ++r) acc[r] = bval;
                int row = mt * 16 + nlo;
                const _Float16* xrow = &sX[row * 64];
                const _Float16* hrow = &sH[row * 64];
                acc = __builtin_amdgcn_wmma_f32_16x16x32_f16(false, frag_a_contig(xrow,      group),
                                                             false, bih0, (short)0, acc, false, false);
                acc = __builtin_amdgcn_wmma_f32_16x16x32_f16(false, frag_a_contig(xrow + 32, group),
                                                             false, bih1, (short)0, acc, false, false);
                acc = __builtin_amdgcn_wmma_f32_16x16x32_f16(false, frag_a_contig(hrow,      group),
                                                             false, bhh0, (short)0, acc, false, false);
                acc = __builtin_amdgcn_wmma_f32_16x16x32_f16(false, frag_a_contig(hrow + 32, group),
                                                             false, bhh1, (short)0, acc, false, false);
#pragma unroll
                for (int r = 0; r < 8; ++r) {
                    int brow = mt * 16 + 8 * group + r;
                    sG[brow * 256 + g0 + nlo] = (_Float16)acc[r];
                }
            }
        }
        __syncthreads();

        // cell update: gate order (torch) i, f, g, o
        _Float16* outp = Sout + (size_t)t * BB * HH;
        for (int i = tid; i < 64 * 64; i += 256) {
            int b = i >> 6, u = i & 63;
            float gi = (float)sG[b * 256 + u];
            float gf = (float)sG[b * 256 + 64 + u];
            float gg = (float)sG[b * 256 + 128 + u];
            float go = (float)sG[b * 256 + 192 + u];
            gi = 1.f / (1.f + __expf(-gi));
            gf = 1.f / (1.f + __expf(-gf));
            go = 1.f / (1.f + __expf(-go));
            gg = tanhf(gg);
            float c = gf * sC[i] + gi * gg;
            sC[i] = c;
            float h = go * tanhf(c);
            sH[i] = (_Float16)h;
            outp[i] = (_Float16)h;
        }
        __syncthreads();
    }
}

// ================= head: logits + log_softmax =================
__global__ __launch_bounds__(64) void head_kernel(const _Float16* __restrict__ hlast, // [B,H]
                                                  const float* __restrict__ ow,       // [H,2]
                                                  const float* __restrict__ ob,       // [2]
                                                  float* __restrict__ out)            // [B,2]
{
    int b = threadIdx.x;
    if (b >= BB) return;
    const _Float16* h = hlast + (size_t)b * HH;
    float l0 = ob[0], l1 = ob[1];
#pragma unroll 8
    for (int u = 0; u < HH; ++u) {
        float hv = (float)h[u];
        l0 += hv * ow[u * 2 + 0];
        l1 += hv * ow[u * 2 + 1];
    }
    float m = fmaxf(l0, l1);
    float z = logf(__expf(l0 - m) + __expf(l1 - m));
    out[b * 2 + 0] = l0 - m - z;
    out[b * 2 + 1] = l1 - m - z;
}

// ================================= host launcher =================================
extern "C" void kernel_launch(void* const* d_in, const int* in_sizes, int n_in,
                              void* d_out, int out_size, void* d_ws, size_t ws_size,
                              hipStream_t stream) {
    (void)in_sizes; (void)n_in; (void)out_size; (void)ws_size;
    const float* input    = (const float*)d_in[0];
    const float* conv1_w  = (const float*)d_in[2];
    const float* conv1_b  = (const float*)d_in[3];
    const float* conv2_w  = (const float*)d_in[4];
    const float* conv2_b  = (const float*)d_in[5];
    const float* bn1_g    = (const float*)d_in[6];
    const float* bn1_b    = (const float*)d_in[7];
    const float* bn1_m    = (const float*)d_in[8];
    const float* bn1_v    = (const float*)d_in[9];
    const float* rconv1_w = (const float*)d_in[10];
    const float* rconv1_b = (const float*)d_in[11];
    const float* bn2_g    = (const float*)d_in[12];
    const float* bn2_b    = (const float*)d_in[13];
    const float* bn2_m    = (const float*)d_in[14];
    const float* bn2_v    = (const float*)d_in[15];
    const float* rconv2_w = (const float*)d_in[16];
    const float* rconv2_b = (const float*)d_in[17];
    const float* w_ih     = (const float*)d_in[18];
    const float* w_hh     = (const float*)d_in[19];
    const float* b_ih     = (const float*)d_in[20];
    const float* b_hh     = (const float*)d_in[21];
    const float* out_w    = (const float*)d_in[22];
    const float* out_b    = (const float*)d_in[23];

    // ---- carve workspace ----
    char* ws = (char*)d_ws;
    size_t off = 0;
    auto carve = [&](size_t bytes) -> char* {
        char* p = ws + off;
        off += (bytes + 255) & ~(size_t)255;
        return p;
    };
    const size_t N_ACT   = (size_t)BB * 64 * LL;          // activation elements
    const int    WRES_N  = 64 * 64 * 11;                  // per res-conv weight count
    const int    WL_N    = 256 * 64;                      // per lstm matrix count
    float*     X32 = (float*)carve(N_ACT * 4);
    float*     Y32 = (float*)carve(N_ACT * 4);
    _Float16*  H16 = (_Float16*)carve(N_ACT * 2);         // channels-last conv input
    _Float16* WR16 = (_Float16*)carve((size_t)2 * NRES * WRES_N * 2);
    _Float16* WL16 = (_Float16*)carve((size_t)2 * NL * WL_N * 2);
    _Float16*   S0 = (_Float16*)carve((size_t)LL * BB * HH * 2);
    _Float16*   S1 = (_Float16*)carve((size_t)LL * BB * HH * 2);

    // ---- weight conversions / repack ----
    {
        int n1 = NRES * WRES_N;
        wrepack_kernel<<<(n1 + 255) / 256, 256, 0, stream>>>(rconv1_w, WR16, NRES);
        wrepack_kernel<<<(n1 + 255) / 256, 256, 0, stream>>>(rconv2_w, WR16 + n1, NRES);
        int n2 = NL * WL_N;
        cvt_kernel<<<(n2 + 255) / 256, 256, 0, stream>>>(w_ih, WL16, n2);
        cvt_kernel<<<(n2 + 255) / 256, 256, 0, stream>>>(w_hh, WL16 + n2, n2);
    }

    // ---- front convs ----
    front_kernel<<<(BB * LL + 255) / 256, 256, 0, stream>>>(input, conv1_w, conv1_b,
                                                            conv2_w, conv2_b, X32);

    // ---- residual blocks ----
    int nAct = (int)N_ACT;
    dim3 cgrid(LL / 64, BB);
    for (int ri = 0; ri < NRES; ++ri) {
        bnrelu_cl_kernel<<<(nAct + 255) / 256, 256, 0, stream>>>(
            X32, bn1_g + ri * 64, bn1_b + ri * 64, bn1_m + ri * 64, bn1_v + ri * 64, H16, nAct);
        conv11_wmma_kernel<<<cgrid, 128, 0, stream>>>(
            H16, WR16 + (size_t)ri * WRES_N, rconv1_b + ri * 64, Y32, 0);
        bnrelu_cl_kernel<<<(nAct + 255) / 256, 256, 0, stream>>>(
            Y32, bn2_g + ri * 64, bn2_b + ri * 64, bn2_m + ri * 64, bn2_v + ri * 64, H16, nAct);
        conv11_wmma_kernel<<<cgrid, 128, 0, stream>>>(
            H16, WR16 + (size_t)(NRES + ri) * WRES_N, rconv2_b + ri * 64, X32, 1);
    }

    // ---- to sequence-first f16 ----
    to_seq_kernel<<<(nAct + 255) / 256, 256, 0, stream>>>(X32, S0);

    // ---- 10 LSTM layers, each resident in one workgroup ----
    for (int layer = 0; layer < NL; ++layer) {
        const _Float16* sin = (layer & 1) ? S1 : S0;
        _Float16*      sout = (layer & 1) ? S0 : S1;
        lstm_layer_kernel<<<1, 256, 0, stream>>>(
            sin, sout,
            WL16 + (size_t)layer * WL_N,
            WL16 + (size_t)(NL + layer) * WL_N,
            b_ih + layer * 256, b_hh + layer * 256);
    }
    // NL=10 (even) -> final output sequence lives in S0
    const _Float16* hlast = S0 + (size_t)(LL - 1) * BB * HH;

    // ---- head ----
    head_kernel<<<1, 64, 0, stream>>>(hlast, out_w, out_b, (float*)d_out);
}
